// Conv1D_MEO_88055419502755
// MI455X (gfx1250) — matrix operand order repaired and checked by reading, not compiled, not run on hardware
//
#include <hip/hip_runtime.h>

// MI455X / gfx1250, wave32. f16 WMMA 16x16x32 (f32 accum) for the batched GEMM,
// double-buffered async global->LDS tile DMA (ASYNCcnt) when the toolchain
// exposes the gfx1250 async builtins; synchronous LDS staging fallback otherwise.
//
// Pipeline:
//  K1 chunk means -> K2 gating/softmax/roll/loss -> K3 bias curve merge
//  K3b per-chunk bias -> K4 (x4) separable mode-32 transforms (in-place)
//  K5 expert mix -> Wmix[64,1024,1024] f16 (tile-resident sources in LDS)
//  K5b x -> f16 (overlays dead rw buffer)
//  K6 batched WMMA GEMM y[b] = xh[b] @ Wmix[b]^T + ebias[b]

typedef __attribute__((ext_vector_type(16))) _Float16 v16h;
typedef __attribute__((ext_vector_type(8)))  _Float16 v8h;
typedef __attribute__((ext_vector_type(8)))  float    v8f;

#define E_   8
#define T_   256
#define IN_  1024
#define OUT_ 1024
#define BN_  64

#if __has_builtin(__builtin_amdgcn_global_load_async_to_lds_b128) && \
    __has_builtin(__builtin_amdgcn_s_wait_asynccnt)
#define USE_ASYNC_LDS 1
#else
#define USE_ASYNC_LDS 0
#endif

#if USE_ASYNC_LDS
typedef int v4i_ __attribute__((vector_size(16)));
#define AS1P(p) ((__attribute__((address_space(1))) v4i_*)(p))
#define AS3P(p) ((__attribute__((address_space(3))) v4i_*)(p))
#endif

// ---------------- workspace layout (bytes) ----------------
static constexpr size_t WS_G    = 0;                          // 64*8*4      = 2 KB
static constexpr size_t WS_M    = 4096;                       // 64*1024*4   = 256 KB
static constexpr size_t WS_RB   = WS_M + 262144;              // 8*1024*4    = 32 KB
static constexpr size_t WS_EB   = WS_RB + 32768;              // 64*1024*4   = 256 KB
static constexpr size_t WS_RW   = (size_t)1 << 20;            // rw' f32 32 MB; later xh f16 32 MB
static constexpr size_t WS_WMIX = WS_RW + ((size_t)32 << 20); // Wmix f16 128 MB

// ---------------- K1: per-chunk feature means ----------------
__global__ __launch_bounds__(256)
void k_chunk_mean(const float* __restrict__ x, float* __restrict__ m) {
  int b = blockIdx.x;
  int d = blockIdx.y * 256 + threadIdx.x;
  const float* p = x + (size_t)b * T_ * IN_ + d;
  float s = 0.f;
#pragma unroll 8
  for (int t = 0; t < T_; ++t) s += p[(size_t)t * IN_];
  m[b * IN_ + d] = s * (1.0f / T_);
}

// ---------------- K2: gating, softmax, roll, loss ----------------
__global__ __launch_bounds__(512)
void k_gate(const float* __restrict__ m, const float* __restrict__ wg,
            float* __restrict__ g, float* __restrict__ loss_out) {
  __shared__ float sl[BN_][E_];
  __shared__ float sg[BN_][E_];
  __shared__ float red[2][E_];
  int t = threadIdx.x;
  int b = t >> 3, e = t & 7;
  float acc = 0.f;
  const float* mb = m + b * IN_;
  for (int i = 0; i < IN_; ++i) acc += mb[i] * wg[i * E_ + e];
  sl[b][e] = acc;
  __syncthreads();
  if (t < BN_) {
    float mx = sl[t][0];
#pragma unroll
    for (int k = 1; k < E_; ++k) mx = fmaxf(mx, sl[t][k]);
    float s = 0.f, ex[E_];
#pragma unroll
    for (int k = 0; k < E_; ++k) { ex[k] = expf(sl[t][k] - mx); s += ex[k]; }
    float inv = 1.f / s;
#pragma unroll
    for (int k = 0; k < E_; ++k) sg[t][k] = ex[k] * inv;
  }
  __syncthreads();
  if (t < BN_) {
    // g[i] = (i % N == 0) ? gates[i] : gates[i-1]   (N = 8 chunks per sample)
    int src = ((t & 7) == 0) ? t : t - 1;
#pragma unroll
    for (int k = 0; k < E_; ++k) g[t * E_ + k] = sg[src][k];
  }
  if (t < E_) {
    float imp = 0.f, ld = 0.f;
    for (int bb = 0; bb < BN_; ++bb) {
      imp += sg[bb][t];
      ld  += (sg[bb][t] > 0.f) ? 1.f : 0.f;
    }
    red[0][t] = imp; red[1][t] = ld;
  }
  __syncthreads();
  if (t == 0) {
    float tot = 0.f;
#pragma unroll
    for (int r = 0; r < 2; ++r) {
      float mean = 0.f;
#pragma unroll
      for (int k = 0; k < E_; ++k) mean += red[r][k];
      mean *= (1.f / E_);
      float var = 0.f;
#pragma unroll
      for (int k = 0; k < E_; ++k) { float d = red[r][k] - mean; var += d * d; }
      var *= (1.f / (E_ - 1));                 // ddof = 1
      tot += var / (mean * mean + 1e-10f);
    }
    loss_out[0] = tot * 1e-5f;
  }
}

// ---------------- K3: bias curve merge (rb_flat[e, i*32+k]) ----------------
__global__ __launch_bounds__(1024)
void k_bias_curve(const float* __restrict__ bias, const float* __restrict__ resb,
                  const float* __restrict__ c1b, const float* __restrict__ c2b,
                  float* __restrict__ rb_flat) {
  int e = blockIdx.x;
  __shared__ float r0[32][33];
  __shared__ float r1[32][33];
  __shared__ float a1[32][33];
  __shared__ float a2[32][33];
  int t = threadIdx.x, i = t >> 5, j = t & 31;
  r0[i][j] = bias[e * 1024 + t] - resb[t];
  a1[i][j] = c1b[e * 1024 + t];
  a2[i][j] = c2b[e * 1024 + t];
  __syncthreads();
  float s = 0.f;                  // rb1[k][j] = sum_i c1b[k][i]*r0[i][j]
#pragma unroll
  for (int q = 0; q < 32; ++q) s += a1[i][q] * r0[q][j];
  r1[i][j] = s;
  __syncthreads();
  s = 0.f;                        // rb2[i][k] = sum_j c2b[k][j]*rb1[i][j]
#pragma unroll
  for (int q = 0; q < 32; ++q) s += a2[j][q] * r1[i][q];
  rb_flat[e * 1024 + i * 32 + j] = s;   // o = i*32 + k
}

// ---------------- K3b: per-chunk merged bias ----------------
__global__ __launch_bounds__(1024)
void k_expert_bias(const float* __restrict__ g, const float* __restrict__ rb,
                   const float* __restrict__ resb, float* __restrict__ eb) {
  int b = blockIdx.x, o = threadIdx.x;
  float s = resb[o];
#pragma unroll
  for (int e = 0; e < E_; ++e) s += g[b * E_ + e] * rb[e * 1024 + o];
  eb[b * 1024 + o] = s;
}

// ---------------- K4: apply 32x32 matrix along one tensor mode ----------------
// out[e, outer, q', inner] = sum_q A[e,q',q] * src[e, outer, q, inner]
// INNER = 1<<sh; each thread owns one full fiber -> in-place safe.
__global__ __launch_bounds__(256)
void k_mode32(const float* __restrict__ src, const float* __restrict__ res,
              const float* __restrict__ mats, float* __restrict__ dst, int sh) {
  int e = blockIdx.y;
  int col = blockIdx.x * 256 + threadIdx.x;          // 0..32767
  int inner = col & ((1 << sh) - 1);
  int outer = col >> sh;
  int OUTER = 32768 >> sh;
  __shared__ float A[32][32];
  for (int k = threadIdx.x; k < 1024; k += 256) A[k >> 5][k & 31] = mats[e * 1024 + k];
  __syncthreads();
  size_t INNER = (size_t)1 << sh;
  size_t base  = (((size_t)e * OUTER + outer) * 32) * INNER + inner;
  size_t rbase = ((size_t)outer * 32) * INNER + inner;
  float v[32];
#pragma unroll
  for (int q = 0; q < 32; ++q) {
    float xv = src[base + (size_t)q * INNER];
    if (res) xv -= res[rbase + (size_t)q * INNER];
    v[q] = xv;
  }
#pragma unroll
  for (int qp = 0; qp < 32; ++qp) {
    float s = 0.f;
#pragma unroll
    for (int q = 0; q < 32; ++q) s += A[qp][q] * v[q];
    dst[base + (size_t)qp * INNER] = s;
  }
}

// ---------------- K5: expert mix -> Wmix f16 (sources read once) ----------------
__global__ __launch_bounds__(256)
void k_mix(const float* __restrict__ rw, const float* __restrict__ resw,
           const float* __restrict__ g, _Float16* __restrict__ wmix) {
  int o0 = blockIdx.x * 32, i0 = blockIdx.y * 32;
  __shared__ float s[9][32][32];                     // 8 experts + residual
  int t = threadIdx.x;
  for (int idx = t; idx < 9 * 1024; idx += 256) {
    int sm = idx >> 10;
    int rc = idx & 1023, r = rc >> 5, c = rc & 31;
    float v = (sm < 8) ? rw[((size_t)sm * OUT_ + o0 + r) * IN_ + i0 + c]
                       : resw[(size_t)(o0 + r) * IN_ + i0 + c];
    s[sm][r][c] = v;
  }
  __syncthreads();
  int r0 = t >> 5, c = t & 31;
  for (int b = 0; b < BN_; ++b) {
    float ge[E_];
#pragma unroll
    for (int e = 0; e < E_; ++e) ge[e] = g[b * E_ + e];
#pragma unroll
    for (int k = 0; k < 4; ++k) {
      int r = r0 + k * 8;
      float v = s[8][r][c];
#pragma unroll
      for (int e = 0; e < E_; ++e) v += ge[e] * s[e][r][c];
      wmix[((size_t)b * OUT_ + o0 + r) * IN_ + i0 + c] = (_Float16)v;
    }
  }
}

// ---------------- K5b: x (f32) -> xh (f16), one pass ----------------
__global__ __launch_bounds__(256)
void k_xhalf(const float* __restrict__ x, _Float16* __restrict__ xh) {
  size_t i = ((size_t)blockIdx.x * 256 + threadIdx.x) * 8;
  float4 a = *reinterpret_cast<const float4*>(x + i);
  float4 b = *reinterpret_cast<const float4*>(x + i + 4);
  v8h h;
  h[0] = (_Float16)a.x; h[1] = (_Float16)a.y; h[2] = (_Float16)a.z; h[3] = (_Float16)a.w;
  h[4] = (_Float16)b.x; h[5] = (_Float16)b.y; h[6] = (_Float16)b.z; h[7] = (_Float16)b.w;
  *reinterpret_cast<v8h*>(xh + i) = h;
}

// ---------------- K6: batched WMMA GEMM  y[b] = xh[b] @ Wmix[b]^T + eb ----------------
#define BM  128
#define BNT 128
#define BK  32
#define LDK 40   // 80-byte rows: multiple of 16 B (aligned b128 dst) + bank de-phasing

__global__ __launch_bounds__(256)
void k_gemm(const _Float16* __restrict__ xh, const _Float16* __restrict__ wmix,
            const float* __restrict__ eb, float* __restrict__ y) {
  int b  = blockIdx.x;
  int m0 = blockIdx.y * BM;
  int n0 = blockIdx.z * BNT;
  __shared__ _Float16 As[2][BM][LDK];
  __shared__ _Float16 Bs[2][BNT][LDK];
  int tid  = threadIdx.x;
  int lane = tid & 31, wid = tid >> 5;
  int wm = wid & 3, wn = wid >> 2;                   // 4 x 2 wave grid
  int l15 = lane & 15, lh = lane >> 4;

  v8f acc[2][4];
  v8f zero = {};
#pragma unroll
  for (int i = 0; i < 2; ++i)
#pragma unroll
    for (int j = 0; j < 4; ++j) acc[i][j] = zero;

  const _Float16* xg = xh   + ((size_t)b * T_   + m0) * IN_;
  const _Float16* wg = wmix + ((size_t)b * OUT_ + n0) * IN_;

  int lc = tid & 3;          // which 16-byte chunk of the 32-half row
  int lr = tid >> 2;         // 0..63 -> rows lr and lr+64
  const int NT = IN_ / BK;   // 32 tiles

  auto issue_tile = [&](int t, int buf) {
    int k0 = t * BK;
#pragma unroll
    for (int rr = 0; rr < 2; ++rr) {
      int row = lr + rr * 64;
#if USE_ASYNC_LDS
      __builtin_amdgcn_global_load_async_to_lds_b128(
          AS1P(xg + (size_t)row * IN_ + k0 + lc * 8), AS3P(&As[buf][row][lc * 8]), 0, 0);
      __builtin_amdgcn_global_load_async_to_lds_b128(
          AS1P(wg + (size_t)row * IN_ + k0 + lc * 8), AS3P(&Bs[buf][row][lc * 8]), 0, 0);
#else
      uint4 ua = *reinterpret_cast<const uint4*>(xg + (size_t)row * IN_ + k0 + lc * 8);
      uint4 ub = *reinterpret_cast<const uint4*>(wg + (size_t)row * IN_ + k0 + lc * 8);
      *reinterpret_cast<uint4*>(&As[buf][row][lc * 8]) = ua;
      *reinterpret_cast<uint4*>(&Bs[buf][row][lc * 8]) = ub;
#endif
    }
    if (t + 2 < NT) {                                // keep L2 warm two tiles ahead
      __builtin_prefetch(xg + (size_t)lr * IN_ + k0 + 2 * BK, 0, 1);
      __builtin_prefetch(wg + (size_t)lr * IN_ + k0 + 2 * BK, 0, 1);
    }
  };

  issue_tile(0, 0);
#if USE_ASYNC_LDS
  __builtin_amdgcn_s_wait_asynccnt(0);
#endif
  __syncthreads();

  for (int t = 0; t < NT; ++t) {
    int cur = t & 1;
    if (t + 1 < NT) issue_tile(t + 1, cur ^ 1);      // DMA next tile during compute

    v16h af[2], bf[4];
    // A-frag: lane m holds k = base..base+7 and base+16..base+23, base = 8*lh
#pragma unroll
    for (int mi = 0; mi < 2; ++mi) {
      int mrow = wm * 32 + mi * 16 + l15;
      v8h lo = *reinterpret_cast<const v8h*>(&As[cur][mrow][lh * 8]);
      v8h hi = *reinterpret_cast<const v8h*>(&As[cur][mrow][lh * 8 + 16]);
#pragma unroll
      for (int q = 0; q < 8; ++q) { af[mi][q] = lo[q]; af[mi][8 + q] = hi[q]; }
    }
    // B-frag: lane n holds k = base..base+7 and base+16..base+23, base = 8*lh, of column n (= W row n)
#pragma unroll
    for (int ni = 0; ni < 4; ++ni) {
      int nrow = wn * 64 + ni * 16 + l15;
      v8h lo = *reinterpret_cast<const v8h*>(&Bs[cur][nrow][lh * 8]);
      v8h hi = *reinterpret_cast<const v8h*>(&Bs[cur][nrow][lh * 8 + 16]);
#pragma unroll
      for (int q = 0; q < 8; ++q) { bf[ni][q] = lo[q]; bf[ni][8 + q] = hi[q]; }
    }
#pragma unroll
    for (int mi = 0; mi < 2; ++mi)
#pragma unroll
      for (int ni = 0; ni < 4; ++ni)
        acc[mi][ni] = __builtin_amdgcn_wmma_f32_16x16x32_f16(
            false, af[mi], false, bf[ni], (short)0, acc[mi][ni], false, false);

    if (t + 1 < NT) {
#if USE_ASYNC_LDS
      __builtin_amdgcn_s_wait_asynccnt(0);           // next tile landed in LDS
#endif
      __syncthreads();                               // everyone done reading cur
    }
  }

  // epilogue: C/D layout VGPR r -> row r + 8*lh, lane&15 -> col
#pragma unroll
  for (int ni = 0; ni < 4; ++ni) {
    int gn = n0 + wn * 64 + ni * 16 + l15;
    float bv = eb[b * OUT_ + gn];
#pragma unroll
    for (int mi = 0; mi < 2; ++mi) {
#pragma unroll
      for (int r = 0; r < 8; ++r) {
        int gm = m0 + wm * 32 + mi * 16 + r + lh * 8;
        y[((size_t)b * T_ + gm) * OUT_ + gn] = acc[mi][ni][r] + bv;
      }
    }
  }
}

// ---------------- launch ----------------
extern "C" void kernel_launch(void* const* d_in, const int* in_sizes, int n_in,
                              void* d_out, int out_size, void* d_ws, size_t ws_size,
                              hipStream_t stream) {
  const float* x     = (const float*)d_in[0];
  const float* wgate = (const float*)d_in[1];
  const float* weight= (const float*)d_in[2];
  const float* bias  = (const float*)d_in[3];
  const float* resw  = (const float*)d_in[4];
  const float* resb  = (const float*)d_in[5];
  const float* c1i   = (const float*)d_in[6];
  const float* c2i   = (const float*)d_in[7];
  const float* c1o   = (const float*)d_in[8];
  const float* c2o   = (const float*)d_in[9];
  const float* c1b   = (const float*)d_in[10];
  const float* c2b   = (const float*)d_in[11];

  float* y    = (float*)d_out;
  float* loss = y + (size_t)BN_ * T_ * OUT_;   // 16,777,216

  char* ws = (char*)d_ws;
  float*    g    = (float*)(ws + WS_G);
  float*    mbuf = (float*)(ws + WS_M);
  float*    rb   = (float*)(ws + WS_RB);
  float*    eb   = (float*)(ws + WS_EB);
  float*    rw   = (float*)(ws + WS_RW);
  _Float16* xhp  = (_Float16*)(ws + WS_RW);    // overlays rw (dead after k_mix)
  _Float16* wmix = (_Float16*)(ws + WS_WMIX);

  k_chunk_mean <<<dim3(64, 4), 256, 0, stream>>>(x, mbuf);
  k_gate       <<<1, 512, 0, stream>>>(mbuf, wgate, g, loss);
  k_bias_curve <<<8, 1024, 0, stream>>>(bias, resb, c1b, c2b, rb);
  k_expert_bias<<<64, 1024, 0, stream>>>(g, rb, resb, eb);
  // separable Kronecker curve transform: m(c2i), l(c1i), k(c2o), j(c1o)
  k_mode32<<<dim3(128, 8), 256, 0, stream>>>(weight, resw, c2i, rw, 0);
  k_mode32<<<dim3(128, 8), 256, 0, stream>>>(rw, nullptr, c1i, rw, 5);
  k_mode32<<<dim3(128, 8), 256, 0, stream>>>(rw, nullptr, c2o, rw, 10);
  k_mode32<<<dim3(128, 8), 256, 0, stream>>>(rw, nullptr, c1o, rw, 15);
  k_mix  <<<dim3(32, 32), 256, 0, stream>>>(rw, resw, g, wmix);
  k_xhalf<<<8192, 256, 0, stream>>>(x, xhp);   // after k_mix: rw region is dead
  k_gemm <<<dim3(64, 2, 8), 256, 0, stream>>>(xhp, wmix, eb, y);
}
